// harmolearn_14104672600184
// MI455X (gfx1250) — matrix operand order repaired and checked by reading, not compiled, not run on hardware
//
#include <hip/hip_runtime.h>

// ---------------------------------------------------------------------------
// Problem constants (from reference)
// ---------------------------------------------------------------------------
#define CH      1024        // channels
#define TLEN    16384       // time length
#define PAD     8           // WINDOW/2
#define NS      16368       // n_steps = TLEN - WINDOW
#define CHUNK   32          // timesteps per scan chunk
#define NCHUNK  512         // ceil(NS / CHUNK), last chunk has 16 steps
#define TILE_W  64          // input tile width (covers j = 0..47 + slack)
#define LDW     65          // LDS row stride for input tile (odd -> conflict free)
#define BRW     34          // LDS row stride for B_ram tile (8B-aligned rows,
                            // 34*delta % 64 != 0 for delta in [1,32) -> conflict free)

typedef __attribute__((ext_vector_type(16))) _Float16 v16h;
typedef __attribute__((ext_vector_type(8)))  float    v8f;

// ---------------------------------------------------------------------------
// Kernel 1: column means of padded input.
// cm[jp] = (jp in [PAD, TLEN+PAD)) ? mean_c inp[c, jp-PAD] : 0
// ---------------------------------------------------------------------------
__global__ __launch_bounds__(256) void colmean_kernel(
    const float* __restrict__ inp, float* __restrict__ cm) {
  const int j = blockIdx.x * 256 + threadIdx.x;   // 0..16383
  float s0 = 0.f, s1 = 0.f, s2 = 0.f, s3 = 0.f;
  #pragma unroll 4
  for (int c = 0; c < CH; c += 4) {
    s0 += inp[(c + 0) * TLEN + j];
    s1 += inp[(c + 1) * TLEN + j];
    s2 += inp[(c + 2) * TLEN + j];
    s3 += inp[(c + 3) * TLEN + j];
  }
  cm[PAD + j] = (s0 + s1 + s2 + s3) * (1.0f / (float)CH);
  if (blockIdx.x == 0 && threadIdx.x < PAD) {
    cm[threadIdx.x] = 0.0f;                 // left padding columns
    cm[TLEN + PAD + threadIdx.x] = 0.0f;    // right padding columns
  }
}

// ---------------------------------------------------------------------------
// Kernel 2: fused WMMA sliding-dot + sequential scan.
// One wave per block; block owns channels [c0, c0+32).
// ---------------------------------------------------------------------------
__global__ __launch_bounds__(32) void scan_kernel(
    const float* __restrict__ inp, const float* __restrict__ cm,
    float* __restrict__ out) {
  __shared__ float lds_in[32 * LDW];   // input tile: row=channel, col=j offset
  __shared__ float lds_cm[48];         // colmean slice for this chunk
  __shared__ float lds_br[32 * BRW];   // B_ram tile: row=channel, col=step
  __shared__ float lds_sp[32 * 33];    // spike tile: row=channel, col=step

  const int lane = threadIdx.x;        // 0..31 (wave32)
  const int c0   = blockIdx.x * 32;    // first channel owned by this block
  const int ncol = lane & 15;
  const int hi   = lane >> 4;          // lane half (ISA K-split)

  // Chunk-invariant band mask for the A (Mt) fragment:
  // element i maps to K = j(i); keep cm[t0+j] iff srow <= j < srow+16.
  float bmask[16];
  {
    const int srow = ncol;
    #pragma unroll
    for (int i = 0; i < 16; ++i) {
      const int j = i + ((i >= 8) ? 8 : 0) + hi * 8;
      bmask[i] = (j >= srow && j < srow + 16) ? 1.0f : 0.0f;
    }
  }

  float w   = 1.0f;                    // w0 = ones
  float mem = 0.0f;                    // mem0 = zeros

  for (int chunk = 0; chunk < NCHUNK; ++chunk) {
    const int tbase = chunk * CHUNK;   // padded-coordinate chunk start

    // ---- (a) stage input tile (padded coords tbase .. tbase+63) ----
    const bool interior = (tbase >= PAD) && (tbase + TILE_W <= TLEN + PAD);
    if (interior) {
      // Unguarded, fully coalesced path (chunks 1..510).
      const float* rowp = inp + (size_t)c0 * TLEN + (tbase - PAD);
      #pragma unroll 8
      for (int r = 0; r < 32; ++r) {
        lds_in[r * LDW + lane]      = rowp[r * TLEN + lane];
        lds_in[r * LDW + lane + 32] = rowp[r * TLEN + lane + 32];
      }
      // Prefetch next chunk's fresh cacheline per row (overlaps the serial
      // recurrence below with the next tile's memory latency).
      if (tbase + 96 <= TLEN + PAD)
        __builtin_prefetch(inp + (size_t)(c0 + lane) * TLEN + (tbase + 64 - PAD), 0, 3);
    } else {
      #pragma unroll 4
      for (int r = 0; r < 32; ++r) {
        #pragma unroll
        for (int g = 0; g < 2; ++g) {
          const int col = g * 32 + lane;
          const int jp  = tbase + col;
          float v = 0.0f;
          if (jp >= PAD && jp < TLEN + PAD)
            v = inp[(c0 + r) * TLEN + (jp - PAD)];
          lds_in[r * LDW + col] = v;
        }
      }
    }
    lds_cm[lane] = cm[tbase + lane];                      // cols 0..31
    if (lane < 16) lds_cm[32 + lane] = cm[tbase + 32 + lane];  // cols 32..47
    __syncthreads();

    // ---- (b) 2 time-tiles x 2 channel-tiles of WMMA ----
    #pragma unroll
    for (int tt = 0; tt < 2; ++tt) {
      const int jb = tt * 16;          // time-tile offset within chunk

      // A fragment: Mt (16 steps x 32 K) = banded colmean.
      // Unconditional LDS loads + multiply by hoisted 0/1 mask (no exec ops).
      v16h a;
      #pragma unroll
      for (int i = 0; i < 16; ++i) {
        const int j = i + ((i >= 8) ? 8 : 0) + hi * 8;
        a[i] = (_Float16)(lds_cm[jb + j] * bmask[i]);
      }

      #pragma unroll
      for (int ct = 0; ct < 2; ++ct) {
        // B fragment: At (32 K x 16 channels).
        v16h b;
        const int row = ct * 16 + ncol;          // channel (local)
        #pragma unroll
        for (int i = 0; i < 16; ++i) {
          const int j = i + ((i >= 8) ? 8 : 0) + hi * 8;
          b[i] = (_Float16)lds_in[row * LDW + jb + j];
        }

        v8f d = {};
        d = __builtin_amdgcn_wmma_f32_16x16x32_f16(
            /*neg_a=*/false, a, /*neg_b=*/false, b,
            /*c_mod=*/(short)0, d, /*reuse_a=*/false, /*reuse_b=*/false);

        // D layout: VGPR r, lane -> (M = r + hi*8, N = lane&15).
        // Channel-major lds_br: a lane's 8 D values are contiguous along s,
        // 8B-aligned base -> merges into ds_store_b64 pairs, conflict free.
        float* dst = &lds_br[(ct * 16 + ncol) * BRW + jb + hi * 8];
        #pragma unroll
        for (int r8 = 0; r8 < 8; ++r8) dst[r8] = d[r8];
      }
    }
    __syncthreads();

    // ---- (c) sequential recurrence: lane = channel c0+lane ----
    // Fixed 32 iterations (state past step NS-1 is dead; stores are guarded).
    #pragma unroll
    for (int s = 0; s < CHUNK; ++s) {
      const float bram = lds_br[lane * BRW + s];
      w = 0.5f * w + 0.5f * bram;
      w = fminf(3.0f, fmaxf(-1.0f, w));          // clip to [W_MIN, W_MAX]
      const float x = lds_in[lane * LDW + s];    // x = inp_pad[c, t]
      const float reset = (mem > 1.0f) ? 1.0f : 0.0f;
      mem = 0.95f * mem + w * x - reset;         // BETA=0.95, THRESHOLD=1
      lds_sp[lane * 33 + s] = (mem > 1.0f) ? 1.0f : 0.0f;
    }
    __syncthreads();

    // ---- (d) coalesced transposed spike write ----
    int smax = NS - tbase;
    if (smax > CHUNK) smax = CHUNK;
    #pragma unroll 8
    for (int r = 0; r < 32; ++r) {
      if (lane < smax)
        out[(c0 + r) * TLEN + tbase + lane] = lds_sp[r * 33 + lane];
    }
    __syncthreads();   // protect LDS reuse next chunk
  }

  // ---- tail: out[:, NS:TLEN) = inp[:, NS:TLEN) ----
  if (lane < 16) {
    #pragma unroll 4
    for (int r = 0; r < 32; ++r) {
      const int idx = (c0 + r) * TLEN + NS + lane;
      out[idx] = inp[idx];
    }
  }
}

// ---------------------------------------------------------------------------
// Launch
// ---------------------------------------------------------------------------
extern "C" void kernel_launch(void* const* d_in, const int* in_sizes, int n_in,
                              void* d_out, int out_size, void* d_ws, size_t ws_size,
                              hipStream_t stream) {
  const float* inp = (const float*)d_in[0];
  float* out = (float*)d_out;
  float* cm  = (float*)d_ws;           // 16400 floats = 65.6 KB of scratch

  colmean_kernel<<<TLEN / 256, 256, 0, stream>>>(inp, cm);
  scan_kernel<<<CH / 32, 32, 0, stream>>>(inp, cm, out);
}
